// HypoDiscovery_21363167331107
// MI455X (gfx1250) — compile-verified, matrix operand order checked by reading, not run on hardware
//
#include <hip/hip_runtime.h>
#include <hip/hip_bf16.h>

typedef __attribute__((ext_vector_type(16))) _Float16 v16h;
typedef __attribute__((ext_vector_type(8)))  _Float16 v8h;
typedef __attribute__((ext_vector_type(8)))  float    v8f;

// Problem sizes (fixed by the reference)
constexpr int kB   = 32768;
constexpr int kD   = 512;
constexpr int kE   = 10;
constexpr int kH   = 128;
constexpr int kL   = 2;
constexpr int kRH1 = 256;
constexpr int kRH2 = 128;
constexpr float kTau = 3.0f;

// ---------------------------------------------------------------------------
// Prep kernels
// ---------------------------------------------------------------------------
__global__ void f32_to_f16_kernel(const float* __restrict__ src,
                                  _Float16* __restrict__ dst, int n) {
  int i = blockIdx.x * blockDim.x + threadIdx.x;
  if (i < n) dst[i] = (_Float16)src[i];
}

// W: [K,N] fp32 (optionally batched) -> WT: [N,K] f16 (B^T, so B columns are
// contiguous in memory for fragment loads).
__global__ void transpose_to_f16_kernel(const float* __restrict__ W,
                                        _Float16* __restrict__ WT,
                                        int K, int N) {
  size_t batch = blockIdx.y;
  const float* w = W + batch * (size_t)K * N;
  _Float16* wt = WT + batch * (size_t)K * N;
  int idx = blockIdx.x * blockDim.x + threadIdx.x;
  if (idx < K * N) {
    int n = idx / K;
    int k = idx - n * K;
    wt[idx] = (_Float16)w[(size_t)k * N + n];
  }
}

// ---------------------------------------------------------------------------
// Fragment helpers.
//
// CDNA5 16-bit A (16x32) lane layout: lane l (0..15) holds row M=l with
// K = {kb..kb+7, kb+16..kb+23}; lanes 16..31 hold the same rows with
// K = {kb+8..kb+15, kb+24..kb+31}.  B mirrors this with column N=l, which is
// a contiguous row of BT.  C/D: VGPR r, lane l -> (M = r + 8*(l>>4), N = l&15).
// ---------------------------------------------------------------------------
__device__ __forceinline__ v16h load_frag_g(const _Float16* p) {
  v8h a = *(const v8h*)(p);       // 16B -> global_load_b128
  v8h b = *(const v8h*)(p + 16);
  v16h r;
#pragma unroll
  for (int i = 0; i < 8; ++i) { r[i] = a[i]; r[i + 8] = b[i]; }
  return r;
}

__device__ __forceinline__ v16h load_frag_lds(const _Float16* p) {
  v8h a = *(const v8h*)(p);       // 16B -> ds_load_b128
  v8h b = *(const v8h*)(p + 16);
  v16h r;
#pragma unroll
  for (int i = 0; i < 8; ++i) { r[i] = a[i]; r[i + 8] = b[i]; }
  return r;
}

__device__ __forceinline__ v8f wmma_f16(v16h a, v16h b, v8f c) {
  return __builtin_amdgcn_wmma_f32_16x16x32_f16(false, a, false, b, (short)0,
                                                c, false, false);
}

// ---------------------------------------------------------------------------
// WMMA GEMM v3: C = relu(A[M,K] * B[K,N] + bias)
//   - block = 256 threads = 8 waves, block tile 256M x 64N
//   - B tile (64 x K, padded rows) staged in LDS once, shared by all waves
//   - each wave computes 32M x 64N (8 accumulators)
//   - k-loop unrolled by 64 with two alternating A buffers (no copy-swap),
//     B fragments pipelined one step ahead, prefetch offsets wrapped instead
//     of guarded (branch-free steady state)
// Requires K % 64 == 0 (true for 512 / 256 / 128). Batched via blockIdx.z.
// ---------------------------------------------------------------------------
template <bool RELU>
__global__ __launch_bounds__(256)
void gemm_wmma_f16(const _Float16* __restrict__ A, int lda, long long aStride,
                   const _Float16* __restrict__ BT, long long bStride,
                   const float* __restrict__ bias, long long biasStride,
                   _Float16* __restrict__ C, int ldc, long long cStride,
                   int K) {
  extern __shared__ _Float16 lds[];   // 64 * (K + 8) halves
  const int tid  = threadIdx.x;
  const int lane = tid & 31;
  const int wave = tid >> 5;
  const int hi   = lane >> 4;
  const int l16  = lane & 15;
  const int ldsRow = K + 8;           // pad 8 halves -> conflict-free b128 reads

  const long long e = blockIdx.z;
  const _Float16* Ae = A + e * aStride;
  const _Float16* Be = BT + e * bStride;
  const float*    be = bias + e * biasStride;
  _Float16*       Ce = C + e * cStride;

  const int m0 = blockIdx.x * 256 + wave * 32;
  const int n0 = blockIdx.y * 64;

  // ---- stage BT tile [64 x K] into LDS (all 256 threads cooperate) ----
  {
    const int chunksPerRow = K >> 3;        // 8 halves per 16B chunk
    const int total = 64 * chunksPerRow;
    for (int c = tid; c < total; c += 256) {
      int n  = c / chunksPerRow;
      int kc = c - n * chunksPerRow;
      v8h v = *(const v8h*)(Be + (size_t)(n0 + n) * K + kc * 8);
      *(v8h*)(&lds[(size_t)n * ldsRow + kc * 8]) = v;
    }
  }
  __syncthreads();

  v8f acc[2][4];
#pragma unroll
  for (int mi = 0; mi < 2; ++mi)
#pragma unroll
    for (int j = 0; j < 4; ++j)
#pragma unroll
      for (int i = 0; i < 8; ++i) acc[mi][j][i] = 0.0f;

  const _Float16* pa0 = Ae + (size_t)(m0 + l16) * lda + 8 * hi;
  const _Float16* pa1 = pa0 + (size_t)16 * lda;

  // Two alternating A buffers (k and k+32) -> no register copy swap.
  v16h aA[2], aB[2];
  aA[0] = load_frag_g(pa0);
  aA[1] = load_frag_g(pa1);
  aB[0] = load_frag_g(pa0 + 32);
  aB[1] = load_frag_g(pa1 + 32);

  const _Float16* lb = &lds[(size_t)l16 * ldsRow + 8 * hi];
  const int jStep = 16 * ldsRow;

  // B pipeline: bcur always holds the fragment for the WMMA about to issue.
  v16h bcur = load_frag_lds(lb);   // (j=0, k=0)

  for (int kb = 0; kb < K; kb += 64) {
    const int kpre = (kb + 64 < K) ? (kb + 64) : 0;  // wrapped prefetch

    // ---- sub-step 0: k = kb, A = aA ----
#pragma unroll
    for (int j = 0; j < 4; ++j) {
      v16h bnxt = (j < 3) ? load_frag_lds(lb + (j + 1) * jStep + kb)
                          : load_frag_lds(lb + kb + 32);
      acc[0][j] = wmma_f16(aA[0], bcur, acc[0][j]);
      acc[1][j] = wmma_f16(aA[1], bcur, acc[1][j]);
      bcur = bnxt;
    }
    aA[0] = load_frag_g(pa0 + kpre);
    aA[1] = load_frag_g(pa1 + kpre);

    // ---- sub-step 1: k = kb+32, A = aB ----
#pragma unroll
    for (int j = 0; j < 4; ++j) {
      v16h bnxt = (j < 3) ? load_frag_lds(lb + (j + 1) * jStep + kb + 32)
                          : load_frag_lds(lb + kpre);  // next outer, j=0
      acc[0][j] = wmma_f16(aB[0], bcur, acc[0][j]);
      acc[1][j] = wmma_f16(aB[1], bcur, acc[1][j]);
      bcur = bnxt;
    }
    aB[0] = load_frag_g(pa0 + kpre + 32);
    aB[1] = load_frag_g(pa1 + kpre + 32);
  }

  // ---- epilogue: bias + relu, store f16 ----
#pragma unroll
  for (int mi = 0; mi < 2; ++mi) {
#pragma unroll
    for (int j = 0; j < 4; ++j) {
      int n = n0 + 16 * j + l16;
      float bv = be[n];
#pragma unroll
      for (int r = 0; r < 8; ++r) {
        int m = m0 + 16 * mi + r + 8 * hi;
        float v = acc[mi][j][r] + bv;
        if (RELU) v = v > 0.0f ? v : 0.0f;
        Ce[(size_t)m * ldc + n] = (_Float16)v;
      }
    }
  }
}

// ---------------------------------------------------------------------------
// Router tail: logits = rh2 @ rw3 + rb3 ; weights = softmax((logits+gumbel)/tau)
// ---------------------------------------------------------------------------
__global__ void router_tail_kernel(const _Float16* __restrict__ rh2,
                                   const float* __restrict__ rw3,  // [RH2, E]
                                   const float* __restrict__ rb3,  // [E]
                                   const float* __restrict__ u,    // [B, E]
                                   float* __restrict__ wout) {     // [B, E]
  int b = blockIdx.x * blockDim.x + threadIdx.x;
  if (b >= kB) return;
  float acc[kE];
#pragma unroll
  for (int e = 0; e < kE; ++e) acc[e] = rb3[e];
  const _Float16* h = rh2 + (size_t)b * kRH2;
  for (int k = 0; k < kRH2; ++k) {
    float hv = (float)h[k];
#pragma unroll
    for (int e = 0; e < kE; ++e) acc[e] += hv * rw3[k * kE + e];
  }
  float mx = -1e30f;
#pragma unroll
  for (int e = 0; e < kE; ++e) {
    float uu = u[(size_t)b * kE + e];
    uu = fminf(fmaxf(uu, 1e-10f), 1.0f);
    float g = -logf(-logf(uu) + 1e-10f);
    acc[e] = (acc[e] + g) / kTau;
    mx = fmaxf(mx, acc[e]);
  }
  float s = 0.0f;
#pragma unroll
  for (int e = 0; e < kE; ++e) { acc[e] = expf(acc[e] - mx); s += acc[e]; }
  float inv = 1.0f / s;
#pragma unroll
  for (int e = 0; e < kE; ++e) wout[(size_t)b * kE + e] = acc[e] * inv;
}

// ---------------------------------------------------------------------------
// zc = h2 @ ew3 + eb3, written straight into the chart_outputs region [E,B,L]
// ---------------------------------------------------------------------------
__global__ void zc_kernel(const _Float16* __restrict__ h2,   // [E][B][H]
                          const float* __restrict__ ew3,     // [E][H][L]
                          const float* __restrict__ eb3,     // [E][L]
                          float* __restrict__ chart) {       // [E][B][L]
  size_t idx = (size_t)blockIdx.x * blockDim.x + threadIdx.x;
  if (idx >= (size_t)kE * kB) return;
  int e = (int)(idx / kB);
  const _Float16* h = h2 + idx * kH;
  const float* w = ew3 + (size_t)e * kH * kL;
  float a0 = eb3[e * kL + 0];
  float a1 = eb3[e * kL + 1];
  for (int k = 0; k < kH; ++k) {
    float hv = (float)h[k];
    a0 += hv * w[k * kL + 0];
    a1 += hv * w[k * kL + 1];
  }
  chart[idx * kL + 0] = a0;
  chart[idx * kL + 1] = a1;
}

// z[b,l] = sum_e weights[b,e] * chart[e,b,l]
__global__ void z_kernel(const float* __restrict__ wght,
                         const float* __restrict__ chart,
                         float* __restrict__ z) {
  int b = blockIdx.x * blockDim.x + threadIdx.x;
  if (b >= kB) return;
  float z0 = 0.0f, z1 = 0.0f;
#pragma unroll
  for (int e = 0; e < kE; ++e) {
    float w = wght[(size_t)b * kE + e];
    z0 += w * chart[((size_t)e * kB + b) * kL + 0];
    z1 += w * chart[((size_t)e * kB + b) * kL + 1];
  }
  z[(size_t)b * kL + 0] = z0;
  z[(size_t)b * kL + 1] = z1;
}

// ---------------------------------------------------------------------------
extern "C" void kernel_launch(void* const* d_in, const int* in_sizes, int n_in,
                              void* d_out, int out_size, void* d_ws, size_t ws_size,
                              hipStream_t stream) {
  (void)in_sizes; (void)n_in; (void)out_size; (void)ws_size;

  const float* x   = (const float*)d_in[0];
  const float* u   = (const float*)d_in[1];
  const float* rw1 = (const float*)d_in[2];
  const float* rb1 = (const float*)d_in[3];
  const float* rw2 = (const float*)d_in[4];
  const float* rb2 = (const float*)d_in[5];
  const float* rw3 = (const float*)d_in[6];
  const float* rb3 = (const float*)d_in[7];
  const float* ew1 = (const float*)d_in[8];
  const float* eb1 = (const float*)d_in[9];
  const float* ew2 = (const float*)d_in[10];
  const float* eb2 = (const float*)d_in[11];
  const float* ew3 = (const float*)d_in[12];
  const float* eb3 = (const float*)d_in[13];

  float* out   = (float*)d_out;
  float* out_z = out;                                  // [B, L]
  float* out_w = out + (size_t)kB * kL;                // [B, E]
  float* out_c = out_w + (size_t)kB * kE;              // [E, B, L]

  // Workspace carve-up (256B aligned)
  char* ws = (char*)d_ws;
  size_t off = 0;
  auto carve = [&](size_t bytes) -> void* {
    void* p = ws + off;
    off += (bytes + 255) & ~(size_t)255;
    return p;
  };
  _Float16* xh   = (_Float16*)carve((size_t)kB * kD * 2);         // x in f16
  _Float16* rw1T = (_Float16*)carve((size_t)kD * kRH1 * 2);       // [RH1, D]
  _Float16* rw2T = (_Float16*)carve((size_t)kRH1 * kRH2 * 2);     // [RH2, RH1]
  _Float16* ew1T = (_Float16*)carve((size_t)kE * kD * kH * 2);    // [E][H][D]
  _Float16* ew2T = (_Float16*)carve((size_t)kE * kH * kH * 2);    // [E][H][H]
  _Float16* rh1  = (_Float16*)carve((size_t)kB * kRH1 * 2);       // [B, RH1]
  _Float16* rh2  = (_Float16*)carve((size_t)kB * kRH2 * 2);       // [B, RH2]
  _Float16* h1   = (_Float16*)carve((size_t)kE * kB * kH * 2);    // [E][B][H]
  _Float16* h2   = (_Float16*)carve((size_t)kE * kB * kH * 2);    // [E][B][H]

  // 1) conversions / weight transposes
  {
    int n = kB * kD;
    f32_to_f16_kernel<<<dim3((n + 255) / 256), 256, 0, stream>>>(x, xh, n);
  }
  transpose_to_f16_kernel<<<dim3((kD * kRH1 + 255) / 256, 1), 256, 0, stream>>>(rw1, rw1T, kD, kRH1);
  transpose_to_f16_kernel<<<dim3((kRH1 * kRH2 + 255) / 256, 1), 256, 0, stream>>>(rw2, rw2T, kRH1, kRH2);
  transpose_to_f16_kernel<<<dim3((kD * kH + 255) / 256, kE), 256, 0, stream>>>(ew1, ew1T, kD, kH);
  transpose_to_f16_kernel<<<dim3((kH * kH + 255) / 256, kE), 256, 0, stream>>>(ew2, ew2T, kH, kH);

  auto ldsBytes = [](int K) { return (size_t)64 * (K + 8) * sizeof(_Float16); };

  // 2) router MLP (two WMMA GEMMs), then logits+gumbel-softmax tail
  gemm_wmma_f16<true><<<dim3(kB / 256, kRH1 / 64, 1), 256, ldsBytes(kD), stream>>>(
      xh, kD, 0, rw1T, 0, rb1, 0, rh1, kRH1, 0, kD);
  gemm_wmma_f16<true><<<dim3(kB / 256, kRH2 / 64, 1), 256, ldsBytes(kRH1), stream>>>(
      rh1, kRH1, 0, rw2T, 0, rb2, 0, rh2, kRH2, 0, kRH1);
  router_tail_kernel<<<dim3(kB / 256), 256, 0, stream>>>(rh2, rw3, rb3, u, out_w);

  // 3) experts, batched over blockIdx.z (A-stride 0 reuses x for all experts)
  gemm_wmma_f16<true><<<dim3(kB / 256, kH / 64, kE), 256, ldsBytes(kD), stream>>>(
      xh, kD, 0,
      ew1T, (long long)kD * kH,
      eb1, kH,
      h1, kH, (long long)kB * kH, kD);
  gemm_wmma_f16<true><<<dim3(kB / 256, kH / 64, kE), 256, ldsBytes(kH), stream>>>(
      h1, kH, (long long)kB * kH,
      ew2T, (long long)kH * kH,
      eb2, kH,
      h2, kH, (long long)kB * kH, kH);

  // 4) zc -> chart_outputs region; z = weights . zc
  zc_kernel<<<dim3((kE * kB + 255) / 256), 256, 0, stream>>>(h2, ew3, eb3, out_c);
  z_kernel<<<dim3(kB / 256), 256, 0, stream>>>(out_w, out_c, out_z);
}
